// SelfAttnHead_128849019493
// MI455X (gfx1250) — matrix operand order repaired and checked by reading, not compiled, not run on hardware
//
#include <hip/hip_runtime.h>
#include <hip/hip_bf16.h>
#include <math.h>

// ---------------------------------------------------------------------------
// Types for CDNA5 WMMA (wave32): v_wmma_f32_16x16x32_bf16
//   A: 16x32 bf16  -> v16bf (8 VGPRs)   B: 32x16 bf16 -> v16bf (8 VGPRs)
//   C/D: 16x16 f32 -> v8f   (8 VGPRs)
// ---------------------------------------------------------------------------
typedef __attribute__((ext_vector_type(4)))  __bf16 v4bf;
typedef __attribute__((ext_vector_type(8)))  __bf16 v8bf;
typedef __attribute__((ext_vector_type(16))) __bf16 v16bf;
typedef __attribute__((ext_vector_type(8)))  float  v8f;

#define WAVE 32

// A-fragment (16x32 bf16, row-major source, ld in elements).
// Layout (ISA 7.12.2): lane<16: row=lane, elems 0-7 = K 0-7, elems 8-15 = K 16-23
//                      lane>=16: row=lane-16, elems 0-7 = K 8-15, elems 8-15 = K 24-31
__device__ __forceinline__ v16bf load_a_frag(const __bf16* __restrict__ A, int lda,
                                             int row0, int k0, int half, int lr) {
    const __bf16* p = A + (size_t)(row0 + lr) * lda + k0 + half * 8;
    v8bf lo = *(const v8bf*)(p);
    v8bf hi = *(const v8bf*)(p + 16);
    return __builtin_shufflevector(lo, hi, 0, 1, 2, 3, 4, 5, 6, 7,
                                   8, 9, 10, 11, 12, 13, 14, 15);
}

// B-fragment (32x16 bf16) built from row-major Bt[N,K] (i.e. computing A @ Bt^T).
// Layout: lane<16: col N = lane, elems 0-7 = K 0-7, elems 8-15 = K 16-23; lane>=16: col N = lane-16, elems 0-7 = K 8-15, elems 8-15 = K 24-31.
__device__ __forceinline__ v16bf load_b_frag(const __bf16* __restrict__ Bt, int ldb,
                                             int col0, int k0, int half, int lr) {
    const __bf16* p = Bt + (size_t)(col0 + lr) * ldb + k0 + half * 8;
    v8bf lo = *(const v8bf*)(p);
    v8bf hi = *(const v8bf*)(p + 16);
    return __builtin_shufflevector(lo, hi, 0, 1, 2, 3, 4, 5, 6, 7,
                                   8, 9, 10, 11, 12, 13, 14, 15);
}

// ---------------------------------------------------------------------------
// Generic NT GEMM: C[M,N] = A[M,K] @ Bt[N,K]^T  (bf16 inputs, f32 accumulate)
// Block: 256 threads = 8 waves arranged 4(M) x 2(N); wave tile 64x64 -> block 256x128.
// EPI 0: C(bf16) = acc + bias[n]
// EPI 1: C(f32)  = (row >= col) ? acc * scale : 0   (multiplicative tril mask)
//        with zero-fill early-out for blocks fully above the diagonal
// EPI 2: C(f32)  = acc
// ---------------------------------------------------------------------------
template <int EPI>
__global__ __launch_bounds__(256) void gemm_nt(
    const __bf16* __restrict__ A, const __bf16* __restrict__ Bt,
    void* __restrict__ Cv, const float* __restrict__ bias,
    int M, int N, int K, int lda, int ldb, int ldc,
    long sA, long sB, long sC, float scale)
{
    const int bz = blockIdx.z;
    A  += (size_t)bz * sA;
    Bt += (size_t)bz * sB;

    const int lane = threadIdx.x & (WAVE - 1);
    const int wave = threadIdx.x / WAVE;
    const int half = lane >> 4;      // 0: lanes 0-15, 1: lanes 16-31
    const int lr   = lane & 15;

    const int mb = blockIdx.y * 256;
    const int nb = blockIdx.x * 128;
    const int wm0 = mb + (wave & 3) * 64;
    const int wn0 = nb + (wave >> 2) * 64;

    if (EPI == 1) {
        // whole block strictly above the diagonal -> masked tile is all zeros
        if (mb + 256 <= nb) {
            float* C = (float*)Cv + (size_t)bz * sC;
            for (int idx = threadIdx.x; idx < 256 * 128; idx += 256) {
                int r = idx >> 7, c = idx & 127;
                C[(size_t)(mb + r) * ldc + (nb + c)] = 0.0f;
            }
            return;
        }
    }

    v8f acc[4][4];
#pragma unroll
    for (int i = 0; i < 4; ++i)
#pragma unroll
        for (int j = 0; j < 4; ++j)
#pragma unroll
            for (int e = 0; e < 8; ++e) acc[i][j][e] = 0.0f;

    for (int k0 = 0; k0 < K; k0 += 32) {
        v16bf af[4], bfm[4];
#pragma unroll
        for (int i = 0; i < 4; ++i) af[i]  = load_a_frag(A,  lda, wm0 + i * 16, k0, half, lr);
#pragma unroll
        for (int j = 0; j < 4; ++j) bfm[j] = load_b_frag(Bt, ldb, wn0 + j * 16, k0, half, lr);
#pragma unroll
        for (int i = 0; i < 4; ++i)
#pragma unroll
            for (int j = 0; j < 4; ++j)
                acc[i][j] = __builtin_amdgcn_wmma_f32_16x16x32_bf16(
                    false, af[i], false, bfm[j], (short)0, acc[i][j], false, false);
    }

    // Epilogue.  C fragment layout: elem e -> row = wm0+i*16 + half*8 + e, col = wn0+j*16 + lr.
#pragma unroll
    for (int i = 0; i < 4; ++i) {
#pragma unroll
        for (int j = 0; j < 4; ++j) {
            const int col = wn0 + j * 16 + lr;
#pragma unroll
            for (int e = 0; e < 8; ++e) {
                const int row = wm0 + i * 16 + half * 8 + e;
                const float val = acc[i][j][e];
                if (EPI == 0) {
                    __bf16* C = (__bf16*)Cv + (size_t)bz * sC;
                    C[(size_t)row * ldc + col] = (__bf16)(val + bias[col]);
                } else if (EPI == 1) {
                    float* C = (float*)Cv + (size_t)bz * sC;
                    C[(size_t)row * ldc + col] = (row >= col) ? val * scale : 0.0f;
                } else {
                    float* C = (float*)Cv + (size_t)bz * sC;
                    C[(size_t)row * ldc + col] = val;
                }
            }
        }
    }
}

// ---------------------------------------------------------------------------
// Prep kernels
// ---------------------------------------------------------------------------
__global__ void cvt_f32_bf16(const float* __restrict__ in, __bf16* __restrict__ out, size_t n) {
    size_t i4 = ((size_t)blockIdx.x * blockDim.x + threadIdx.x) * 4;
    if (i4 + 3 < n) {
        float4 v = *(const float4*)(in + i4);
        v4bf o;
        o[0] = (__bf16)v.x; o[1] = (__bf16)v.y; o[2] = (__bf16)v.z; o[3] = (__bf16)v.w;
        *(v4bf*)(out + i4) = o;
    } else {
        for (size_t k = i4; k < n; ++k) out[k] = (__bf16)in[k];
    }
}

// in: f32 [rows, cols] row-major  ->  out: bf16 [cols, rows]
__global__ void transpose_cvt_f32_bf16(const float* __restrict__ in, __bf16* __restrict__ out,
                                       int rows, int cols) {
    __shared__ float tile[32][33];
    const int c0 = blockIdx.x * 32, r0 = blockIdx.y * 32;
    for (int i = threadIdx.y; i < 32; i += 8)
        tile[i][threadIdx.x] = in[(size_t)(r0 + i) * cols + c0 + threadIdx.x];
    __syncthreads();
    for (int i = threadIdx.y; i < 32; i += 8)
        out[(size_t)(c0 + i) * rows + r0 + threadIdx.x] = (__bf16)tile[threadIdx.x][i];
}

// batched: in bf16 [rows, cols] -> out bf16 [cols, rows], blockIdx.z = batch
__global__ void transpose_bf16(const __bf16* __restrict__ in, __bf16* __restrict__ out,
                               int rows, int cols) {
    __shared__ __bf16 tile[32][33];
    const size_t ib = (size_t)blockIdx.z * rows * cols;
    const int c0 = blockIdx.x * 32, r0 = blockIdx.y * 32;
    for (int i = threadIdx.y; i < 32; i += 8)
        tile[i][threadIdx.x] = in[ib + (size_t)(r0 + i) * cols + c0 + threadIdx.x];
    __syncthreads();
    for (int i = threadIdx.y; i < 32; i += 8)
        out[ib + (size_t)(c0 + i) * rows + r0 + threadIdx.x] = tile[threadIdx.x][i];
}

// ---------------------------------------------------------------------------
// Column softmax over the QUERY axis (axis=1 of [B, q, k]); masked zeros stay
// in the distribution (exp(0)=1), per the reference.  One thread per k column,
// online max/sum; k is the fast axis so all accesses are coalesced.
// ---------------------------------------------------------------------------
__global__ void softmax_col(const float* __restrict__ S, __bf16* __restrict__ attn, int T) {
    const int b = blockIdx.y;
    const int k = blockIdx.x * blockDim.x + threadIdx.x;
    const float* col = S + (size_t)b * T * T + k;
    float m = -INFINITY, d = 0.0f;
    for (int q = 0; q < T; ++q) {
        float x = col[(size_t)q * T];
        float nm = fmaxf(m, x);
        d = d * __expf(m - nm) + __expf(x - nm);
        m = nm;
    }
    const float inv = 1.0f / d;
    __bf16* ocol = attn + (size_t)b * T * T + k;
    for (int q = 0; q < T; ++q) {
        float x = col[(size_t)q * T];
        ocol[(size_t)q * T] = (__bf16)(__expf(x - m) * inv);
    }
}

// ---------------------------------------------------------------------------
// Host-side orchestration
// ---------------------------------------------------------------------------
extern "C" void kernel_launch(void* const* d_in, const int* in_sizes, int n_in,
                              void* d_out, int out_size, void* d_ws, size_t ws_size,
                              hipStream_t stream) {
    (void)in_sizes; (void)n_in; (void)out_size; (void)ws_size;

    const int Bn = 4, T = 2048, E = 1024, H = 2048;
    const int BT = Bn * T;
    const float scale = 0.022097086912079608f;  // 1/sqrt(2048)

    const float* x  = (const float*)d_in[0];
    const float* Wq = (const float*)d_in[1];
    const float* bq = (const float*)d_in[2];
    const float* Wv = (const float*)d_in[3];
    const float* bv = (const float*)d_in[4];
    float* out = (float*)d_out;

    // Workspace layout (~184 MiB)
    char* ws = (char*)d_ws;
    size_t off = 0;
    auto carve = [&](size_t bytes) -> void* {
        void* p = ws + off;
        off += (bytes + 255) & ~(size_t)255;
        return p;
    };
    __bf16* xb    = (__bf16*)carve((size_t)BT * E * 2);   // x in bf16
    __bf16* WqT   = (__bf16*)carve((size_t)H * E * 2);    // Wq^T bf16 [H,E]
    __bf16* WvT   = (__bf16*)carve((size_t)H * E * 2);    // Wv^T bf16 [H,E]
    __bf16* qb    = (__bf16*)carve((size_t)BT * H * 2);   // q bf16 [B*T,H]
    __bf16* vb    = (__bf16*)carve((size_t)BT * H * 2);   // v bf16 [B*T,H]; reused as attn bf16
    __bf16* vT    = (__bf16*)carve((size_t)Bn * H * T * 2); // v^T bf16 [B,H,T]
    float*  S     = (float*)carve((size_t)Bn * T * T * 4);  // masked scaled scores f32
    __bf16* attnb = vb;  // safe alias: vb is dead after the transpose

    // --- prep: conversions / weight transposes ---
    {
        size_t n = (size_t)BT * E;
        cvt_f32_bf16<<<dim3((unsigned)((n / 4 + 255) / 256)), 256, 0, stream>>>(x, xb, n);
    }
    transpose_cvt_f32_bf16<<<dim3(H / 32, E / 32), dim3(32, 8), 0, stream>>>(Wq, WqT, E, H);
    transpose_cvt_f32_bf16<<<dim3(H / 32, E / 32), dim3(32, 8), 0, stream>>>(Wv, WvT, E, H);

    // --- projections: q = x@Wq + bq, v = x@Wv + bv ---
    gemm_nt<0><<<dim3(H / 128, BT / 256, 1), 256, 0, stream>>>(
        xb, WqT, qb, bq, BT, H, E, E, E, H, 0, 0, 0, 0.0f);
    gemm_nt<0><<<dim3(H / 128, BT / 256, 1), 256, 0, stream>>>(
        xb, WvT, vb, bv, BT, H, E, E, E, H, 0, 0, 0, 0.0f);

    // --- v^T per batch for the attn@v B-operand ---
    transpose_bf16<<<dim3(H / 32, T / 32, Bn), dim3(32, 8), 0, stream>>>(vb, vT, T, H);

    // --- S = tril(q @ q^T / sqrt(H)) (multiplicative mask), per batch ---
    gemm_nt<1><<<dim3(T / 128, T / 256, Bn), 256, 0, stream>>>(
        qb, qb, S, nullptr, T, T, H, H, H, T,
        (long)T * H, (long)T * H, (long)T * T, scale);

    // --- softmax over the query axis (columns of [q,k]) ---
    softmax_col<<<dim3(T / 256, Bn), 256, 0, stream>>>(S, attnb, T);

    // --- out = attn @ v  (contract over k using v^T rows) ---
    gemm_nt<2><<<dim3(H / 128, T / 256, Bn), 256, 0, stream>>>(
        attnb, vT, out, nullptr, T, H, T, T, T, H,
        (long)T * T, (long)H * T, (long)T * H, 1.0f);
}
